// GIN_5970004541461
// MI455X (gfx1250) — matrix operand order repaired and checked by reading, not compiled, not run on hardware
//
#include <hip/hip_runtime.h>
#include <hip/hip_bf16.h>
#include <math.h>

typedef __attribute__((ext_vector_type(16))) _Float16 v16h;
typedef __attribute__((ext_vector_type(8)))  _Float16 v8h;
typedef __attribute__((ext_vector_type(4)))  _Float16 v4h;
typedef __attribute__((ext_vector_type(8)))  float    v8f;

#define D 64
#define C 40
#define TILE_M 128
#define APAD 72   // padded LDS row stride in halves: 144B rows, keeps 16B align

// ---------------------------------------------------------------------------
// Edge aggregation: Agg[dst] += X[src].  16 lanes per edge, float4 per lane.
// ---------------------------------------------------------------------------
__global__ __launch_bounds__(256) void agg_kernel(const int* __restrict__ ei,
                                                  const float* __restrict__ X,
                                                  float* __restrict__ Agg,
                                                  int E) {
  long long t = (long long)blockIdx.x * blockDim.x + threadIdx.x;
  long long e = t >> 4;
  if (e >= E) return;
  int f4 = (int)(t & 15);
  int s = ei[e];
  int d = ei[(long long)E + e];
  const float4 v = ((const float4*)(X + (size_t)s * D))[f4];
  float* dst = Agg + (size_t)d * D + f4 * 4;
  atomicAdd(dst + 0, v.x);
  atomicAdd(dst + 1, v.y);
  atomicAdd(dst + 2, v.z);
  atomicAdd(dst + 3, v.w);
}

// ---------------------------------------------------------------------------
// Fused GIN MLP: O = relu( relu(A @ W1 + b1) @ W2 + b2 )
// A: [N x 64] f32. Weights staged into LDS in WMMA B-fragment-major layout so
// each lane's B operand is one aligned 32-byte LDS read. A fragments are two
// aligned 16-byte LDS reads. 8 waves/block, 16 rows/wave, 128 rows/block.
// ---------------------------------------------------------------------------
__global__ __launch_bounds__(256) void mlp_kernel(const float* __restrict__ A,
                                                  float* __restrict__ O,
                                                  const float* __restrict__ w1,
                                                  const float* __restrict__ b1,
                                                  const float* __restrict__ w2,
                                                  const float* __restrict__ b2,
                                                  int N) {
  // Fragment-major weights: [k-step][n-tile][lane] -> v16h (32B each, 8KB per W)
  __shared__ v16h Wf1[2][4][32];
  __shared__ v16h Wf2[2][4][32];
  __shared__ _Float16 As[TILE_M * APAD];
  __shared__ _Float16 Ts[TILE_M * APAD];
  __shared__ float b1s[D];
  __shared__ float b2s[D];

  const int tid = threadIdx.x;

  // ---- Stage weights as per-lane WMMA B fragments (one fragment per thread).
  {
    const int ks  = tid >> 7;         // 0..1  (K block of 32)
    const int nt  = (tid >> 5) & 3;   // 0..3  (N tile of 16)
    const int ln  = tid & 31;
    const int rlo = ln & 15;
    const int hi  = ln >> 4;
    const int col = nt * 16 + rlo;
    const int kbb = ks * 32 + hi * 16;
    v16h f1, f2;
    #pragma unroll
    for (int j = 0; j < 16; ++j) {
      f1[j] = (_Float16)w1[(kbb + j) * D + col];
      f2[j] = (_Float16)w2[(kbb + j) * D + col];
    }
    Wf1[ks][nt][ln] = f1;
    Wf2[ks][nt][ln] = f2;
  }
  if (tid < D) {
    b1s[tid] = b1[tid];
    b2s[tid] = b2[tid];
  }

  // ---- Stage A tile: float4 global loads -> packed v4h LDS stores.
  const int rowBase = blockIdx.x * TILE_M;
  for (int i = tid; i < TILE_M * (D / 4); i += 256) {
    const int r  = i >> 4;       // row in tile
    const int c4 = i & 15;       // float4 slot
    const int gr = rowBase + r;
    float4 v = make_float4(0.f, 0.f, 0.f, 0.f);
    if (gr < N) v = ((const float4*)(A + (size_t)gr * D))[c4];
    v4h h;
    h[0] = (_Float16)v.x; h[1] = (_Float16)v.y;
    h[2] = (_Float16)v.z; h[3] = (_Float16)v.w;
    *(v4h*)&As[r * APAD + c4 * 4] = h;
  }
  __syncthreads();

  const int wave = tid >> 5;       // 0..7
  const int lane = tid & 31;
  const int m0   = wave * 16;      // row tile inside block tile
  const int rlo  = lane & 15;
  const int hi   = lane >> 4;      // half-wave
  const int arow = (m0 + rlo) * APAD;

  v8f acc[4];

  // ---------------- GEMM1: T = relu(As @ W1 + b1) ----------------
  #pragma unroll
  for (int nt = 0; nt < 4; ++nt)
    #pragma unroll
    for (int i = 0; i < 8; ++i) acc[nt][i] = 0.0f;

  #pragma unroll
  for (int ks = 0; ks < 2; ++ks) {
    const int kb = ks * 32 + hi * 16;
    const v8h alo = *(const v8h*)&As[arow + kb];
    const v8h ahi = *(const v8h*)&As[arow + kb + 8];
    const v16h a = __builtin_shufflevector(alo, ahi, 0, 1, 2, 3, 4, 5, 6, 7,
                                           8, 9, 10, 11, 12, 13, 14, 15);
    #pragma unroll
    for (int nt = 0; nt < 4; ++nt) {
      const v16h b = Wf1[ks][nt][lane];
      acc[nt] = __builtin_amdgcn_wmma_f32_16x16x32_f16(
          false, a, false, b, (short)0, acc[nt], false, false);
    }
  }

  // bias + relu -> staged f16 intermediate tile (C layout -> row-major)
  #pragma unroll
  for (int nt = 0; nt < 4; ++nt) {
    const int col = nt * 16 + rlo;
    const float bb = b1s[col];
    #pragma unroll
    for (int i = 0; i < 8; ++i) {
      const int row = m0 + i + hi * 8;
      float v = acc[nt][i] + bb;
      Ts[row * APAD + col] = (_Float16)(v > 0.0f ? v : 0.0f);
    }
  }
  __syncthreads();

  // ---------------- GEMM2: O = relu(T @ W2 + b2) ----------------
  #pragma unroll
  for (int nt = 0; nt < 4; ++nt)
    #pragma unroll
    for (int i = 0; i < 8; ++i) acc[nt][i] = 0.0f;

  #pragma unroll
  for (int ks = 0; ks < 2; ++ks) {
    const int kb = ks * 32 + hi * 16;
    const v8h alo = *(const v8h*)&Ts[arow + kb];
    const v8h ahi = *(const v8h*)&Ts[arow + kb + 8];
    const v16h a = __builtin_shufflevector(alo, ahi, 0, 1, 2, 3, 4, 5, 6, 7,
                                           8, 9, 10, 11, 12, 13, 14, 15);
    #pragma unroll
    for (int nt = 0; nt < 4; ++nt) {
      const v16h b = Wf2[ks][nt][lane];
      acc[nt] = __builtin_amdgcn_wmma_f32_16x16x32_f16(
          false, a, false, b, (short)0, acc[nt], false, false);
    }
  }

  #pragma unroll
  for (int nt = 0; nt < 4; ++nt) {
    const int col = nt * 16 + rlo;
    const float bb = b2s[col];
    #pragma unroll
    for (int i = 0; i < 8; ++i) {
      const int r = rowBase + m0 + i + hi * 8;
      if (r < N) {
        float v = acc[nt][i] + bb;
        O[(size_t)r * D + col] = (v > 0.0f ? v : 0.0f);
      }
    }
  }
}

// ---------------------------------------------------------------------------
// BatchNorm stats: sums[c] = sum_n X[n][c], sums[64+c] = sum_n X[n][c]^2
// ---------------------------------------------------------------------------
__global__ __launch_bounds__(256) void stats_kernel(const float* __restrict__ X,
                                                    float* __restrict__ sums,
                                                    int N) {
  __shared__ float rs[256];
  __shared__ float rq[256];
  const int col = threadIdx.x & 63;
  const int rg  = threadIdx.x >> 6;  // 0..3
  float s = 0.0f, q = 0.0f;
  for (long long r = (long long)blockIdx.x * 4 + rg; r < N;
       r += (long long)gridDim.x * 4) {
    float v = X[r * D + col];
    s += v;
    q += v * v;
  }
  rs[threadIdx.x] = s;
  rq[threadIdx.x] = q;
  __syncthreads();
  if (rg == 0) {
    s = rs[col] + rs[64 + col] + rs[128 + col] + rs[192 + col];
    q = rq[col] + rq[64 + col] + rq[128 + col] + rq[192 + col];
    atomicAdd(&sums[col], s);
    atomicAdd(&sums[D + col], q);
  }
}

// ---------------------------------------------------------------------------
// BatchNorm apply: Y = (X - mean) * rsqrt(var + eps) * g + be
// ---------------------------------------------------------------------------
__global__ __launch_bounds__(256) void bn_kernel(const float* __restrict__ X,
                                                 float* __restrict__ Y,
                                                 const float* __restrict__ sums,
                                                 const float* __restrict__ g,
                                                 const float* __restrict__ be,
                                                 int N) {
  long long i = (long long)blockIdx.x * blockDim.x + threadIdx.x;
  long long total = (long long)N * D;
  if (i >= total) return;
  int c = (int)(i & 63);
  float invN = 1.0f / (float)N;
  float mean = sums[c] * invN;
  float var  = sums[D + c] * invN - mean * mean;
  float v = (X[i] - mean) * rsqrtf(var + 1e-5f) * g[c] + be[c];
  Y[i] = v;
}

// ---------------------------------------------------------------------------
// Final FC (64x40) + log_softmax. One wave32 per row; lane l owns cols l and
// 32+l (latter valid when l<8). Wave reductions via __shfl_xor.
// ---------------------------------------------------------------------------
__global__ __launch_bounds__(256) void fc_kernel(const float* __restrict__ H,
                                                 const float* __restrict__ fcw,
                                                 const float* __restrict__ fcb,
                                                 float* __restrict__ out,
                                                 int N) {
  __shared__ float Ws[D * C];
  __shared__ float bs[C];
  for (int i = threadIdx.x; i < D * C; i += 256) Ws[i] = fcw[i];
  if (threadIdx.x < C) bs[threadIdx.x] = fcb[threadIdx.x];
  __syncthreads();

  const int wave = threadIdx.x >> 5;
  const int lane = threadIdx.x & 31;
  const int c1 = lane;
  const int c2 = 32 + (lane & 7);
  const bool v2 = lane < 8;

  for (long long row = (long long)blockIdx.x * 8 + wave; row < N;
       row += (long long)gridDim.x * 8) {
    float a0 = bs[c1];
    float a1 = bs[c2];
    const float* hrow = H + row * D;
    #pragma unroll 8
    for (int k = 0; k < D; ++k) {
      float h = hrow[k];
      a0 = fmaf(h, Ws[k * C + c1], a0);
      a1 = fmaf(h, Ws[k * C + c2], a1);
    }
    float m = fmaxf(a0, v2 ? a1 : -INFINITY);
    #pragma unroll
    for (int off = 16; off > 0; off >>= 1)
      m = fmaxf(m, __shfl_xor(m, off, 32));
    float e = __expf(a0 - m) + (v2 ? __expf(a1 - m) : 0.0f);
    #pragma unroll
    for (int off = 16; off > 0; off >>= 1)
      e += __shfl_xor(e, off, 32);
    float lse = __logf(e);
    float* orow = out + row * C;
    orow[c1] = a0 - m - lse;
    if (v2) orow[c2] = a1 - m - lse;
  }
}

// ---------------------------------------------------------------------------
extern "C" void kernel_launch(void* const* d_in, const int* in_sizes, int n_in,
                              void* d_out, int out_size, void* d_ws, size_t ws_size,
                              hipStream_t stream) {
  const float* x   = (const float*)d_in[0];
  const int*   ei  = (const int*)d_in[1];
  const float* w1a = (const float*)d_in[2];
  const float* b1a = (const float*)d_in[3];
  const float* w1b = (const float*)d_in[4];
  const float* b1b = (const float*)d_in[5];
  const float* w2a = (const float*)d_in[6];
  const float* b2a = (const float*)d_in[7];
  const float* w2b = (const float*)d_in[8];
  const float* b2b = (const float*)d_in[9];
  const float* w3a = (const float*)d_in[10];
  const float* b3a = (const float*)d_in[11];
  const float* w3b = (const float*)d_in[12];
  const float* b3b = (const float*)d_in[13];
  const float* g1  = (const float*)d_in[14];
  const float* be1 = (const float*)d_in[15];
  const float* g2  = (const float*)d_in[16];
  const float* be2 = (const float*)d_in[17];
  const float* fcw = (const float*)d_in[18];
  const float* fcb = (const float*)d_in[19];
  float* out = (float*)d_out;

  const int N = in_sizes[0] / D;
  const int E = in_sizes[1] / 2;
  const size_t feats = (size_t)N * D;
  const size_t featBytes = feats * sizeof(float);

  float* P     = (float*)d_ws;
  float* Q     = P + feats;
  float* stats = Q + feats;

  const int aggBlocks = (int)(((long long)E * 16 + 255) / 256);
  const int mlpBlocks = (N + TILE_M - 1) / TILE_M;
  const int bnBlocks  = (int)((feats + 255) / 256);

  // ---------------- Layer 1 ----------------
  hipMemcpyAsync(P, x, featBytes, hipMemcpyDeviceToDevice, stream);
  agg_kernel<<<aggBlocks, 256, 0, stream>>>(ei, x, P, E);
  mlp_kernel<<<mlpBlocks, 256, 0, stream>>>(P, Q, w1a, b1a, w1b, b1b, N);
  hipMemsetAsync(stats, 0, 2 * D * sizeof(float), stream);
  stats_kernel<<<1024, 256, 0, stream>>>(Q, stats, N);
  bn_kernel<<<bnBlocks, 256, 0, stream>>>(Q, P, stats, g1, be1, N);

  // ---------------- Layer 2 ----------------
  hipMemcpyAsync(Q, P, featBytes, hipMemcpyDeviceToDevice, stream);
  agg_kernel<<<aggBlocks, 256, 0, stream>>>(ei, P, Q, E);
  mlp_kernel<<<mlpBlocks, 256, 0, stream>>>(Q, P, w2a, b2a, w2b, b2b, N);
  hipMemsetAsync(stats, 0, 2 * D * sizeof(float), stream);
  stats_kernel<<<1024, 256, 0, stream>>>(P, stats, N);
  bn_kernel<<<bnBlocks, 256, 0, stream>>>(P, Q, stats, g2, be2, N);

  // ---------------- Layer 3 ----------------
  hipMemcpyAsync(P, Q, featBytes, hipMemcpyDeviceToDevice, stream);
  agg_kernel<<<aggBlocks, 256, 0, stream>>>(ei, Q, P, E);
  mlp_kernel<<<mlpBlocks, 256, 0, stream>>>(P, Q, w3a, b3a, w3b, b3b, N);

  // ---------------- FC + log_softmax ----------------
  fc_kernel<<<4096, 256, 0, stream>>>(Q, fcw, fcb, out, N);
}
